// StackPropagationSlu_49271864820391
// MI455X (gfx1250) — matrix operand order repaired and checked by reading, not compiled, not run on hardware
//
#include <hip/hip_runtime.h>
#include <hip/hip_bf16.h>
#include <stdint.h>

// ---------------------------------------------------------------------------
// Model dims
// ---------------------------------------------------------------------------
constexpr int kB  = 1024;   // batch
constexpr int kT  = 64;     // seq len
constexpr int kE  = 256;    // embedding dim
constexpr int kH  = 512;    // LSTM units per direction
constexpr int kAH = 1024;   // attention q/k dim
constexpr int kAO = 128;    // attention v/out dim
constexpr int kDU = 64;     // decoder units
constexpr int kNI = 64;     // intent size
constexpr int kNS = 128;    // slot size
constexpr int kEW = 2 * kH + kAO;  // e width = 1152

typedef _Float16 h16v __attribute__((ext_vector_type(16)));
typedef _Float16 h8v  __attribute__((ext_vector_type(8)));
typedef float    f8v  __attribute__((ext_vector_type(8)));

__device__ __forceinline__ h16v cat8(h8v lo, h8v hi) {
  return __builtin_shufflevector(lo, hi, 0, 1, 2, 3, 4, 5, 6, 7,
                                         8, 9, 10, 11, 12, 13, 14, 15);
}

// ---------------------------------------------------------------------------
// Generic batched GEMM: C[M,N] = scale * (A @ B) + bias
//   A = row-concat of two segments: A0 [M,K0] (lda0), A1 [M,K1] (lda1);
//   segments 32-aligned in K. B supplied TRANSPOSED: BT[n*ldb + k] (f16).
//   REQUIRES: M % 32 == 0, N % 64 == 0 (true for all uses here).
//   Output f32 (Cf) or f16 (Ch). blockIdx.z = batch with strides bsA/bsB/bsC
//   (batched mode only uses segment A0).
// Wave32 WMMA f32_16x16x32_f16; each wave owns a 32x64 tile (2x4 accumulator
// grid: each B fragment is reused for two WMMAs from registers), block = 8
// waves covering 256x64. Branch-free inner loop:
//   load 2 A frags -> 4 x (load B frag, 2 x v_wmma).
// ---------------------------------------------------------------------------
__global__ void __launch_bounds__(256) gemm_wmma_f16(
    const _Float16* __restrict__ A0, long lda0, int K0,
    const _Float16* __restrict__ A1, long lda1, int K1,
    const _Float16* __restrict__ BT, long ldb,
    const float* __restrict__ bias,
    float* __restrict__ Cf, _Float16* __restrict__ Ch, long ldc,
    int M, int N, float scale,
    long bsA, long bsB, long bsC)
{
  const int wid  = threadIdx.x >> 5;
  const int lane = threadIdx.x & 31;
  A0 += (long)blockIdx.z * bsA;
  BT += (long)blockIdx.z * bsB;
  const long coff = (long)blockIdx.z * bsC;

  const int row = blockIdx.y * 256 + wid * 32;
  if (row >= M) return;                         // wave-uniform
  const int col0 = blockIdx.x * 64;

  const int m   = row + (lane & 15);
  const int kbA = (lane & 16) ? 8  : 0;         // A: lanes16-31 hold K+8..15 / +24..31
  const int kbB = (lane & 16) ? 8  : 0;         // B: lanes16-31 hold K+8..15 / +24..31
  const int nn  = lane & 15;

  f8v acc[2][4];
  #pragma unroll
  for (int r = 0; r < 2; ++r)
    #pragma unroll
    for (int c = 0; c < 4; ++c)
      #pragma unroll
      for (int i = 0; i < 8; ++i) acc[r][c][i] = 0.0f;

  const _Float16* arow0 = A0 + (long)m * lda0 + kbA;          // rows m, m+16
  const _Float16* arow1 = (K1 > 0) ? (A1 + (long)m * lda1 + kbA) : nullptr;
  const _Float16* brow  = BT + (long)(col0 + nn) * ldb + kbB;

  // ---- segment 0 ----
  for (int k0 = 0; k0 < K0; k0 += 32) {
    const _Float16* ap0 = arow0 + k0;
    const h16v a0 = cat8(*(const h8v*)(ap0),
                         *(const h8v*)(ap0 + 16));
    const h16v a1 = cat8(*(const h8v*)(ap0 + 16 * lda0),
                         *(const h8v*)(ap0 + 16 * lda0 + 16));
    __builtin_prefetch(ap0 + 32, 0, 3);              // -> global_prefetch_b8 (near)
    __builtin_prefetch(ap0 + 16 * lda0 + 32, 0, 3);
    #pragma unroll
    for (int ct = 0; ct < 4; ++ct) {
      const _Float16* bp = brow + (long)(ct * 16) * ldb + k0;
      const h16v b = cat8(*(const h8v*)(bp), *(const h8v*)(bp + 16));
      acc[0][ct] = __builtin_amdgcn_wmma_f32_16x16x32_f16(
          false, a0, false, b, (short)0, acc[0][ct], false, false);
      acc[1][ct] = __builtin_amdgcn_wmma_f32_16x16x32_f16(
          false, a1, false, b, (short)0, acc[1][ct], false, false);
    }
  }
  // ---- segment 1 ----
  for (int k1 = 0; k1 < K1; k1 += 32) {
    const int kg = K0 + k1;                          // global K for B
    const _Float16* ap1 = arow1 + k1;
    const h16v a0 = cat8(*(const h8v*)(ap1),
                         *(const h8v*)(ap1 + 16));
    const h16v a1 = cat8(*(const h8v*)(ap1 + 16 * lda1),
                         *(const h8v*)(ap1 + 16 * lda1 + 16));
    #pragma unroll
    for (int ct = 0; ct < 4; ++ct) {
      const _Float16* bp = brow + (long)(ct * 16) * ldb + kg;
      const h16v b = cat8(*(const h8v*)(bp), *(const h8v*)(bp + 16));
      acc[0][ct] = __builtin_amdgcn_wmma_f32_16x16x32_f16(
          false, a0, false, b, (short)0, acc[0][ct], false, false);
      acc[1][ct] = __builtin_amdgcn_wmma_f32_16x16x32_f16(
          false, a1, false, b, (short)0, acc[1][ct], false, false);
    }
  }

  // ---- store (N % 64 == 0: no column guards) ----
  const int mhi = (lane & 16) ? 8 : 0;
  if (Cf) {
    #pragma unroll
    for (int r = 0; r < 2; ++r)
      #pragma unroll
      for (int ct = 0; ct < 4; ++ct) {
        const int n = col0 + ct * 16 + nn;
        const float bval = bias ? bias[n] : 0.0f;
        #pragma unroll
        for (int i = 0; i < 8; ++i)
          Cf[coff + (long)(row + r * 16 + mhi + i) * ldc + n] =
              acc[r][ct][i] * scale + bval;
      }
  } else {
    #pragma unroll
    for (int r = 0; r < 2; ++r)
      #pragma unroll
      for (int ct = 0; ct < 4; ++ct) {
        const int n = col0 + ct * 16 + nn;
        const float bval = bias ? bias[n] : 0.0f;
        #pragma unroll
        for (int i = 0; i < 8; ++i)
          Ch[coff + (long)(row + r * 16 + mhi + i) * ldc + n] =
              (_Float16)(acc[r][ct][i] * scale + bval);
      }
  }
}

// ---------------------------------------------------------------------------
// LSTM cell (keras gate order i,f,g,o). z: [B,4U] f32. c: [B,U] f32 in/out.
// h written as f16 at hout[b*ldh + u].
// ---------------------------------------------------------------------------
__global__ void lstm_cell(const float* __restrict__ z, float* __restrict__ c,
                          _Float16* __restrict__ hout, long ldh, int U, int B)
{
  const int idx = blockIdx.x * blockDim.x + threadIdx.x;
  if (idx >= B * U) return;
  const int b = idx / U, u = idx % U;
  const float* zr = z + (long)b * 4 * U;
  const float zi = zr[u], zf = zr[U + u], zg = zr[2 * U + u], zo = zr[3 * U + u];
  const float si = 1.0f / (1.0f + expf(-zi));
  const float sf = 1.0f / (1.0f + expf(-zf));
  const float so = 1.0f / (1.0f + expf(-zo));
  const float cn = sf * c[idx] + si * tanhf(zg);
  c[idx] = cn;
  hout[(long)b * ldh + u] = (_Float16)(so * tanhf(cn));
}

// ---------------------------------------------------------------------------
// Attention: softmax over tk (64) + context = P @ V (64x128), one wave per
// (b, tq). Writes context f16 into e[:, 1024:1152].
// ---------------------------------------------------------------------------
__global__ void __launch_bounds__(256) attn_softmax_ctx(
    const float* __restrict__ scores, const _Float16* __restrict__ vbuf,
    _Float16* __restrict__ e, int B)
{
  __shared__ float pP[8][64];
  const int wid  = threadIdx.x >> 5;
  const int lane = threadIdx.x & 31;
  const int idx  = blockIdx.x * 8 + wid;        // = b*64 + tq
  const int b  = idx >> 6;
  const int tq = idx & 63;

  const float* s = scores + ((long)b * 64 + tq) * 64;
  float s0 = s[lane], s1 = s[lane + 32];
  float mx = fmaxf(s0, s1);
  #pragma unroll
  for (int off = 16; off; off >>= 1) mx = fmaxf(mx, __shfl_xor(mx, off));
  float e0 = expf(s0 - mx), e1 = expf(s1 - mx);
  float sm = e0 + e1;
  #pragma unroll
  for (int off = 16; off; off >>= 1) sm += __shfl_xor(sm, off);
  const float inv = 1.0f / sm;
  pP[wid][lane]      = e0 * inv;
  pP[wid][lane + 32] = e1 * inv;
  __syncthreads();

  float acc0 = 0.f, acc1 = 0.f, acc2 = 0.f, acc3 = 0.f;
  const int d = lane * 4;
  for (int tk = 0; tk < 64; ++tk) {
    const float pw = pP[wid][tk];
    const _Float16* vp = vbuf + ((long)tk * B + b) * kAO + d;
    acc0 += pw * (float)vp[0];
    acc1 += pw * (float)vp[1];
    acc2 += pw * (float)vp[2];
    acc3 += pw * (float)vp[3];
  }
  _Float16* ep = e + ((long)tq * B + b) * kEW + 2 * kH + d;
  ep[0] = (_Float16)acc0; ep[1] = (_Float16)acc1;
  ep[2] = (_Float16)acc2; ep[3] = (_Float16)acc3;
}

// ---------------------------------------------------------------------------
// Intent head: softmax(64) -> yi output (f32), pint f16 into dynS[:, 0:64],
// feedback pi = int_emb[argmax] f16 into dynI[:, 0:8]. One wave per b.
// ---------------------------------------------------------------------------
__global__ void __launch_bounds__(256) intent_head(
    const float* __restrict__ logits, float* __restrict__ yi,
    const _Float16* __restrict__ intEmb, _Float16* __restrict__ dynI,
    _Float16* __restrict__ dynS, int t)
{
  const int wid  = threadIdx.x >> 5;
  const int lane = threadIdx.x & 31;
  const int b = blockIdx.x * 8 + wid;
  const float* lr = logits + (long)b * kNI;
  float v0 = lr[lane], v1 = lr[lane + 32];

  float mv; int mi;
  if (v0 >= v1) { mv = v0; mi = lane; } else { mv = v1; mi = lane + 32; }
  #pragma unroll
  for (int off = 16; off; off >>= 1) {
    float ov = __shfl_xor(mv, off);
    int   oi = __shfl_xor(mi, off);
    if (ov > mv || (ov == mv && oi < mi)) { mv = ov; mi = oi; }
  }
  float e0 = expf(v0 - mv), e1 = expf(v1 - mv);
  float sm = e0 + e1;
  #pragma unroll
  for (int off = 16; off; off >>= 1) sm += __shfl_xor(sm, off);
  const float inv = 1.0f / sm;
  const float p0 = e0 * inv, p1 = e1 * inv;

  float* yo = yi + ((long)b * kT + t) * kNI;
  yo[lane] = p0; yo[lane + 32] = p1;
  dynS[(long)b * 160 + lane]      = (_Float16)p0;
  dynS[(long)b * 160 + lane + 32] = (_Float16)p1;
  if (lane < 8) dynI[(long)b * 96 + lane] = intEmb[mi * 8 + lane];
}

// ---------------------------------------------------------------------------
// Slot head: softmax(128) -> ys output (f32), feedback ps = slot_emb[argmax]
// f16 into dynS[:, 64:96]. One wave per b.
// ---------------------------------------------------------------------------
__global__ void __launch_bounds__(256) slot_head(
    const float* __restrict__ logits, float* __restrict__ ys,
    const _Float16* __restrict__ slotEmb, _Float16* __restrict__ dynS, int t)
{
  const int wid  = threadIdx.x >> 5;
  const int lane = threadIdx.x & 31;
  const int b = blockIdx.x * 8 + wid;
  const float* lr = logits + (long)b * kNS;
  float v[4];
  #pragma unroll
  for (int j = 0; j < 4; ++j) v[j] = lr[lane + 32 * j];

  float mv = v[0]; int mi = lane;
  #pragma unroll
  for (int j = 1; j < 4; ++j)
    if (v[j] > mv) { mv = v[j]; mi = lane + 32 * j; }
  #pragma unroll
  for (int off = 16; off; off >>= 1) {
    float ov = __shfl_xor(mv, off);
    int   oi = __shfl_xor(mi, off);
    if (ov > mv || (ov == mv && oi < mi)) { mv = ov; mi = oi; }
  }
  float ex[4]; float sm = 0.f;
  #pragma unroll
  for (int j = 0; j < 4; ++j) { ex[j] = expf(v[j] - mv); sm += ex[j]; }
  #pragma unroll
  for (int off = 16; off; off >>= 1) sm += __shfl_xor(sm, off);
  const float inv = 1.0f / sm;

  float* yo = ys + ((long)b * kT + t) * kNS;
  #pragma unroll
  for (int j = 0; j < 4; ++j) yo[lane + 32 * j] = ex[j] * inv;
  dynS[(long)b * 160 + 64 + lane] = slotEmb[mi * 32 + lane];
}

// ---------------------------------------------------------------------------
// Embedding gather f32 -> f16, time-major X[t*B + b, :]
// ---------------------------------------------------------------------------
__global__ void embed_gather(const int* __restrict__ ids,
                             const float* __restrict__ emb,
                             _Float16* __restrict__ X)
{
  const long tid = (long)blockIdx.x * blockDim.x + threadIdx.x;
  const long total = (long)kB * kT * kE / 4;
  if (tid >= total) return;
  const long el = tid * 4;
  const long c  = el % kE;
  const long r  = el / kE;            // r = t*B + b
  const int t = (int)(r / kB);
  const int b = (int)(r % kB);
  const int id = ids[(long)b * kT + t];
  const float* s = emb + (long)id * kE + c;
  _Float16* d = X + r * kE + c;
  d[0] = (_Float16)s[0]; d[1] = (_Float16)s[1];
  d[2] = (_Float16)s[2]; d[3] = (_Float16)s[3];
}

// ---------------------------------------------------------------------------
// Weight prep: WT[n*ldwt + koff + k] = (f16) W[k*N + n]
// ---------------------------------------------------------------------------
__global__ void conv_transpose(const float* __restrict__ W, int K, int N,
                               _Float16* __restrict__ WT, long ldwt, long koff)
{
  const long tid = (long)blockIdx.x * blockDim.x + threadIdx.x;
  if (tid >= (long)K * N) return;
  const long n = tid / K, k = tid % K;
  WT[n * ldwt + koff + k] = (_Float16)W[k * (long)N + n];
}

__global__ void conv_copy(const float* __restrict__ src,
                          _Float16* __restrict__ dst, int n)
{
  const int i = blockIdx.x * blockDim.x + threadIdx.x;
  if (i < n) dst[i] = (_Float16)src[i];
}

// ---------------------------------------------------------------------------
// Host orchestration
// ---------------------------------------------------------------------------
extern "C" void kernel_launch(void* const* d_in, const int* in_sizes, int n_in,
                              void* d_out, int out_size, void* d_ws, size_t ws_size,
                              hipStream_t stream)
{
  (void)in_sizes; (void)n_in; (void)out_size; (void)ws_size;
  const int*   ids   = (const int*)d_in[0];
  const float* emb   = (const float*)d_in[1];
  const float* Wk_f  = (const float*)d_in[2];
  const float* Wr_f  = (const float*)d_in[3];
  const float* b_f   = (const float*)d_in[4];
  const float* Wk_b  = (const float*)d_in[5];
  const float* Wr_b  = (const float*)d_in[6];
  const float* b_b   = (const float*)d_in[7];
  const float* Wq    = (const float*)d_in[8];
  const float* bq    = (const float*)d_in[9];
  const float* Wka   = (const float*)d_in[10];
  const float* bka   = (const float*)d_in[11];
  const float* Wv    = (const float*)d_in[12];
  const float* bvv   = (const float*)d_in[13];
  const float* Wki   = (const float*)d_in[14];
  const float* Wri   = (const float*)d_in[15];
  const float* bi_c  = (const float*)d_in[16];
  const float* Wks   = (const float*)d_in[17];
  const float* Wrs   = (const float*)d_in[18];
  const float* bs_c  = (const float*)d_in[19];
  const float* Wdi   = (const float*)d_in[20];
  const float* bdi   = (const float*)d_in[21];
  const float* Wds   = (const float*)d_in[22];
  const float* bds   = (const float*)d_in[23];
  const float* intE  = (const float*)d_in[24];
  const float* slotE = (const float*)d_in[25];

  float* yi = (float*)d_out;
  float* ys = yi + (size_t)kB * kT * kNI;

  uint8_t* wp = (uint8_t*)d_ws;
  auto alloc = [&](size_t bytes) -> void* {
    void* r = (void*)wp;
    wp += (bytes + 255) & ~(size_t)255;
    return r;
  };

  _Float16* X     = (_Float16*)alloc((size_t)kT * kB * kE * 2);
  _Float16* e     = (_Float16*)alloc((size_t)kT * kB * kEW * 2);
  _Float16* qb    = (_Float16*)alloc((size_t)kT * kB * kAH * 2);
  _Float16* kb    = (_Float16*)alloc((size_t)kT * kB * kAH * 2);
  _Float16* vb    = (_Float16*)alloc((size_t)kT * kB * kAO * 2);
  float*    scor  = (float*)   alloc((size_t)kB * kT * kT * 4);
  float*    zenc  = (float*)   alloc((size_t)kB * 4 * kH * 4);
  float*    cf    = (float*)   alloc((size_t)kB * kH * 4);
  float*    cbk   = (float*)   alloc((size_t)kB * kH * 4);
  _Float16* zeroh = (_Float16*)alloc((size_t)kB * kH * 2);
  _Float16* WcfT  = (_Float16*)alloc((size_t)(4 * kH) * 768 * 2);
  _Float16* WcbT  = (_Float16*)alloc((size_t)(4 * kH) * 768 * 2);
  _Float16* WqT   = (_Float16*)alloc((size_t)kAH * 1024 * 2);
  _Float16* WkaT  = (_Float16*)alloc((size_t)kAH * 1024 * 2);
  _Float16* WvT   = (_Float16*)alloc((size_t)kAO * 1024 * 2);
  _Float16* WkiT  = (_Float16*)alloc((size_t)256 * 1248 * 2);
  _Float16* WksT  = (_Float16*)alloc((size_t)256 * 1312 * 2);
  _Float16* WdiT  = (_Float16*)alloc((size_t)kNI * 64 * 2);
  _Float16* WdsT  = (_Float16*)alloc((size_t)kNS * 64 * 2);
  _Float16* iE16  = (_Float16*)alloc((size_t)kNI * 8 * 2);
  _Float16* sE16  = (_Float16*)alloc((size_t)kNS * 32 * 2);
  float*    zi    = (float*)   alloc((size_t)kB * 256 * 4);
  float*    zs    = (float*)   alloc((size_t)kB * 256 * 4);
  float*    li    = (float*)   alloc((size_t)kB * kNI * 4);
  float*    ls    = (float*)   alloc((size_t)kB * kNS * 4);
  _Float16* dynI  = (_Float16*)alloc((size_t)kB * 96 * 2);   // [pi(8) pad(24) | ih(64)]
  _Float16* dynS  = (_Float16*)alloc((size_t)kB * 160 * 2);  // [pint(64) | ps(32) | sh(64)]
  float*    ic    = (float*)   alloc((size_t)kB * kDU * 4);
  float*    sc    = (float*)   alloc((size_t)kB * kDU * 4);

  // zero-init (capture-safe memset nodes)
  hipMemsetAsync(zeroh, 0, (size_t)kB * kH * 2, stream);
  hipMemsetAsync(cf,   0, (size_t)kB * kH * 4, stream);
  hipMemsetAsync(cbk,  0, (size_t)kB * kH * 4, stream);
  hipMemsetAsync(WkiT, 0, (size_t)256 * 1248 * 2, stream);
  hipMemsetAsync(WksT, 0, (size_t)256 * 1312 * 2, stream);
  hipMemsetAsync(dynI, 0, (size_t)kB * 96 * 2, stream);
  hipMemsetAsync(dynS, 0, (size_t)kB * 160 * 2, stream);
  hipMemsetAsync(ic,   0, (size_t)kB * kDU * 4, stream);
  hipMemsetAsync(sc,   0, (size_t)kB * kDU * 4, stream);

  auto convT = [&](const float* W, int K, int N, _Float16* WT, long ld, long koff) {
    long tot = (long)K * N;
    conv_transpose<<<(unsigned)((tot + 255) / 256), 256, 0, stream>>>(W, K, N, WT, ld, koff);
  };
  // combined recurrent weights [Wk ; Wr]^T for K-concat A = [x_t | h_prev]
  convT(Wk_f, kE, 4 * kH, WcfT, 768, 0);
  convT(Wr_f, kH, 4 * kH, WcfT, 768, kE);
  convT(Wk_b, kE, 4 * kH, WcbT, 768, 0);
  convT(Wr_b, kH, 4 * kH, WcbT, 768, kE);
  convT(Wq,  2 * kH, kAH, WqT, 1024, 0);
  convT(Wka, 2 * kH, kAH, WkaT, 1024, 0);
  convT(Wv,  2 * kH, kAO, WvT, 1024, 0);
  // decoder combined: [Wki(1160) | pad(24) | Wri(64)] => K=1248
  convT(Wki, 1160, 4 * kDU, WkiT, 1248, 0);
  convT(Wri, kDU,  4 * kDU, WkiT, 1248, 1184);
  // [Wks(1248) | Wrs(64)] => K=1312
  convT(Wks, 1248, 4 * kDU, WksT, 1312, 0);
  convT(Wrs, kDU,  4 * kDU, WksT, 1312, 1248);
  convT(Wdi, kDU, kNI, WdiT, 64, 0);
  convT(Wds, kDU, kNS, WdsT, 64, 0);
  conv_copy<<<(kNI * 8 + 255) / 256, 256, 0, stream>>>(intE, iE16, kNI * 8);
  conv_copy<<<(kNS * 32 + 255) / 256, 256, 0, stream>>>(slotE, sE16, kNS * 32);

  // embedding gather
  {
    long tot = (long)kB * kT * kE / 4;
    embed_gather<<<(unsigned)((tot + 255) / 256), 256, 0, stream>>>(ids, emb, X);
  }

  auto gemm = [&](const _Float16* A0, long lda0, int K0,
                  const _Float16* A1, long lda1, int K1,
                  const _Float16* BTm, long ldb, const float* bias,
                  float* Cf, _Float16* Ch, long ldc, int M, int N, float scale,
                  long bsA, long bsB, long bsC, int batches) {
    dim3 g((unsigned)(N / 64), (unsigned)((M + 255) / 256), (unsigned)batches);
    gemm_wmma_f16<<<g, dim3(256), 0, stream>>>(A0, lda0, K0, A1, lda1, K1, BTm, ldb,
                                               bias, Cf, Ch, ldc, M, N, scale,
                                               bsA, bsB, bsC);
  };

  // ---- forward LSTM (h written into e[:, 0:512]) ----
  for (int t = 0; t < kT; ++t) {
    const _Float16* hp = (t == 0) ? zeroh : (e + (size_t)(t - 1) * kB * kEW);
    const long ldah = (t == 0) ? kH : kEW;
    gemm(X + (size_t)t * kB * kE, kE, kE, hp, ldah, kH,
         WcfT, 768, b_f, zenc, nullptr, 4 * kH, kB, 4 * kH, 1.0f, 0, 0, 0, 1);
    lstm_cell<<<(kB * kH + 255) / 256, 256, 0, stream>>>(
        zenc, cf, e + (size_t)t * kB * kEW, kEW, kH, kB);
  }
  // ---- backward LSTM (h written into e[:, 512:1024]) ----
  for (int t = kT - 1; t >= 0; --t) {
    const _Float16* hp = (t == kT - 1) ? zeroh : (e + (size_t)(t + 1) * kB * kEW + kH);
    const long ldah = (t == kT - 1) ? kH : kEW;
    gemm(X + (size_t)t * kB * kE, kE, kE, hp, ldah, kH,
         WcbT, 768, b_b, zenc, nullptr, 4 * kH, kB, 4 * kH, 1.0f, 0, 0, 0, 1);
    lstm_cell<<<(kB * kH + 255) / 256, 256, 0, stream>>>(
        zenc, cbk, e + (size_t)t * kB * kEW + kH, kEW, kH, kB);
  }

  // ---- q, k, v projections (A = e[:, 0:1024], time-major rows) ----
  gemm(e, kEW, 2 * kH, nullptr, 0, 0, WqT,  1024, bq,  nullptr, qb, kAH,
       kT * kB, kAH, 1.0f, 0, 0, 0, 1);
  gemm(e, kEW, 2 * kH, nullptr, 0, 0, WkaT, 1024, bka, nullptr, kb, kAH,
       kT * kB, kAH, 1.0f, 0, 0, 0, 1);
  gemm(e, kEW, 2 * kH, nullptr, 0, 0, WvT,  1024, bvv, nullptr, vb, kAO,
       kT * kB, kAO, 1.0f, 0, 0, 0, 1);

  // ---- attention scores: per-batch [64x1024]@[1024x64], scaled 1/32 ----
  gemm(qb, (long)kB * kAH, kAH, nullptr, 0, 0, kb, (long)kB * kAH, nullptr,
       scor, nullptr, kT, kT, kT, 0.03125f,
       /*bsA*/ kAH, /*bsB*/ kAH, /*bsC*/ (long)kT * kT, kB);

  // ---- softmax + context -> e[:, 1024:1152] ----
  attn_softmax_ctx<<<kB * kT / 8, 256, 0, stream>>>(scor, vb, e, kB);

  // ---- decoder scan ----
  for (int t = 0; t < kT; ++t) {
    const _Float16* et = e + (size_t)t * kB * kEW;
    // intent cell: z = [e_t | pi | ih] @ [Wki;Wri] + b
    gemm(et, kEW, kEW, dynI, 96, 96, WkiT, 1248, bi_c,
         zi, nullptr, 4 * kDU, kB, 4 * kDU, 1.0f, 0, 0, 0, 1);
    lstm_cell<<<(kB * kDU + 255) / 256, 256, 0, stream>>>(zi, ic, dynI + 32, 96, kDU, kB);
    // intent logits + head
    gemm(dynI + 32, 96, kDU, nullptr, 0, 0, WdiT, 64, bdi,
         li, nullptr, kNI, kB, kNI, 1.0f, 0, 0, 0, 1);
    intent_head<<<kB / 8, 256, 0, stream>>>(li, yi, iE16, dynI, dynS, t);
    // slot cell: z = [e_t | pint | ps | sh] @ [Wks;Wrs] + b
    gemm(et, kEW, kEW, dynS, 160, 160, WksT, 1312, bs_c,
         zs, nullptr, 4 * kDU, kB, 4 * kDU, 1.0f, 0, 0, 0, 1);
    lstm_cell<<<(kB * kDU + 255) / 256, 256, 0, stream>>>(zs, sc, dynS + 96, 160, kDU, kB);
    // slot logits + head
    gemm(dynS + 96, 160, kDU, nullptr, 0, 0, WdsT, 64, bds,
         ls, nullptr, kNS, kB, kNS, 1.0f, 0, 0, 0, 1);
    slot_head<<<kB / 8, 256, 0, stream>>>(ls, ys, sE16, dynS, t);
  }
}
